// Decoder_91190745629213
// MI455X (gfx1250) — matrix operand order repaired and checked by reading, not compiled, not run on hardware
//
#include <hip/hip_runtime.h>
#include <hip/hip_bf16.h>
#include <cstddef>

typedef _Float16 f16;
typedef __attribute__((ext_vector_type(16))) _Float16 v16h;
typedef __attribute__((ext_vector_type(8)))  _Float16 v8h;
typedef __attribute__((ext_vector_type(8)))  float    v8f;

// ---------------- problem constants ----------------
#define FRAMES   1200
#define N_NODES  11201
#define NPAD     11232           // 351 * 32  (multiple of 32 for MT=2 tiling)
#define DIM      512
#define NEDGE    100000
#define NREL     5
#define BATCH    32
#define OUTW     1712            // 1200 + 512

// =====================================================================
// f16 WMMA GEMM:  C[M x N] = act(A[M x K] * Bt[N x K]^T + bias)
// A row-major (lda = K), Bt row-major (ldb = K)  -> both load contiguous.
//
// Block = 256 threads = 8 waves. Wave tile = (16*MT) x 64.
//   MT=2: 4 A-loads + 8 B-loads feed 8 WMMAs per K-step (1.5 b128/WMMA),
//         8 independent accumulator chains hide the WMMA RAW window.
//   MT=1: for skinny M=32 GEMMs (MLP head / final projection).
//
// Register layouts per the CDNA5 ISA:
//   A 16x32 f16 : lanes 0-15 (M=lane)     hold K {0..7, 16..23}
//                 lanes 16-31 (M=lane-16) hold K {8..15, 24..31}
//   B 32x16 f16 : lane n<16  -> col n,    K {0..7, 16..23}
//                 lane n>=16 -> col n-16, K {8..15, 24..31}
//   C 16x16 f32 : VGPR g, lanes 0-15 -> (M=g, N=lane); lanes 16-31 -> (M=g+8)
// ACT: 0 = none, 1 = relu, 2 = tanh.  Ch: optional f16 mirror of C.
// M must be a multiple of 16*MT (padded); per-wave M-guard is wave-uniform.
// =====================================================================
template <int ACT, int MT>
__global__ void wmma_gemm_kernel(const f16* __restrict__ A,
                                 const f16* __restrict__ Bt,
                                 float* __restrict__ C,
                                 f16*   __restrict__ Ch,
                                 const float* __restrict__ bias,
                                 int M, int N, int K, int ldc)
{
    const int lane = threadIdx.x & 31;
    const int wave = threadIdx.x >> 5;                 // 0..7
    const int hlf  = lane >> 4;                        // 0 or 1
    const int ls   = lane & 15;
    const int m0   = (blockIdx.x * 8 + wave) * (16 * MT);
    const int n0   = blockIdx.y * 64;
    if (m0 >= M) return;                               // wave-uniform: EXEC stays full

    const f16* arow = A  + (size_t)(m0 + ls) * K + hlf * 8;   // ISA A layout
    const f16* brow = Bt + (size_t)(n0 + ls) * K + hlf * 8;   // ISA B layout

    v8f acc[MT][4] = {};
    for (int k = 0; k < K; k += 32) {
        v16h a[MT];
#pragma unroll
        for (int p = 0; p < MT; ++p) {
            const f16* ap = arow + (size_t)p * 16 * K + k;
            v8h a_lo = *(const v8h*)(ap);          // K +{0..7}   (hlf0) / +{8..15}  (hlf1)
            v8h a_hi = *(const v8h*)(ap + 16);     // K +{16..23} (hlf0) / +{24..31} (hlf1)
#pragma unroll
            for (int i = 0; i < 8; ++i) { a[p][i] = a_lo[i]; a[p][i + 8] = a_hi[i]; }
        }

#pragma unroll
        for (int j = 0; j < 4; ++j) {
            const f16* bj = brow + (size_t)j * 16 * K + k;
            v8h b_lo = *(const v8h*)(bj);          // K +{0..7}   (hlf0) / +{8..15}  (hlf1)
            v8h b_hi = *(const v8h*)(bj + 16);     // K +{16..23} (hlf0) / +{24..31} (hlf1)
            v16h b;
#pragma unroll
            for (int i = 0; i < 8; ++i) { b[i] = b_lo[i]; b[i + 8] = b_hi[i]; }
#pragma unroll
            for (int p = 0; p < MT; ++p) {
                acc[p][j] = __builtin_amdgcn_wmma_f32_16x16x32_f16(
                                false, a[p], false, b, (short)0, acc[p][j], false, false);
            }
        }
    }

#pragma unroll
    for (int p = 0; p < MT; ++p) {
#pragma unroll
        for (int j = 0; j < 4; ++j) {
            const int n = n0 + j * 16 + ls;
            if (n >= N) continue;
            const float bv = bias ? bias[n] : 0.0f;
#pragma unroll
            for (int g = 0; g < 8; ++g) {
                const int m = m0 + p * 16 + g + hlf * 8;
                if (m >= M) continue;
                float v = acc[p][j][g] + bv;
                if (ACT == 1) v = v > 0.0f ? v : 0.0f;
                else if (ACT == 2) v = tanhf(v);
                C[(size_t)m * ldc + n] = v;
                if (Ch) Ch[(size_t)m * ldc + n] = (f16)v;
            }
        }
    }
}

// ---------------- helper kernels ----------------

// Build x = concat(frame_emb[:1200], role_emb[fe_ids]) padded to NPAD rows,
// in both f32 and f16.
__global__ void build_x_kernel(const float* __restrict__ frame_emb,
                               const float* __restrict__ role_emb,
                               const int*   __restrict__ fe_ids,
                               float* __restrict__ x32, f16* __restrict__ xh)
{
    size_t idx = (size_t)blockIdx.x * blockDim.x + threadIdx.x;
    if (idx >= (size_t)NPAD * DIM) return;
    int row = (int)(idx >> 9);
    int col = (int)(idx & (DIM - 1));
    float v = 0.0f;
    if (row < FRAMES)        v = frame_emb[(size_t)row * DIM + col];
    else if (row < N_NODES)  v = role_emb[(size_t)fe_ids[row - FRAMES] * DIM + col];
    x32[idx] = v;
    xh[idx]  = (f16)v;
}

__global__ void zero_kernel(float* __restrict__ p, size_t n)
{
    size_t idx = (size_t)blockIdx.x * blockDim.x + threadIdx.x;
    if (idx < n) p[idx] = 0.0f;
}

// Transposing f32 -> f16 convert:  Wt[n*K + k] = (f16) W[k*N + n]
__global__ void convT_kernel(const float* __restrict__ W, f16* __restrict__ Wt,
                             int K, int N)
{
    size_t idx = (size_t)blockIdx.x * blockDim.x + threadIdx.x;
    if (idx >= (size_t)K * N) return;
    int k = (int)(idx / N);
    int n = (int)(idx % N);
    Wt[(size_t)n * K + k] = (f16)W[idx];
}

__global__ void conv_kernel(const float* __restrict__ src, f16* __restrict__ dst, size_t n)
{
    size_t idx = (size_t)blockIdx.x * blockDim.x + threadIdx.x;
    if (idx < n) dst[idx] = (f16)src[idx];
}

// One block (128 threads) per edge: acc[rows[e]] += vals[e] * xl[cols[e]]
__global__ void edge_scatter_kernel(const float* __restrict__ xl,
                                    const int*   __restrict__ rows,
                                    const int*   __restrict__ cols,
                                    const float* __restrict__ vals,
                                    float* __restrict__ acc)
{
    const int e = blockIdx.x;
    const int r = rows[e];
    const int c = cols[e];
    const float v = vals[e];
    const float4* src = (const float4*)(xl + (size_t)c * DIM);
    float*        dst = acc + (size_t)r * DIM;
    const int i = threadIdx.x;          // 0..127 -> 512 floats
    float4 m = src[i];
    unsafeAtomicAdd(dst + 4 * i + 0, v * m.x);
    unsafeAtomicAdd(dst + 4 * i + 1, v * m.y);
    unsafeAtomicAdd(dst + 4 * i + 2, v * m.z);
    unsafeAtomicAdd(dst + 4 * i + 3, v * m.w);
}

// x32/xh = tanh(acc), elementwise over NPAD*DIM (pad rows stay 0: tanh(0)=0)
__global__ void tanh_kernel(const float* __restrict__ acc,
                            float* __restrict__ x32, f16* __restrict__ xh)
{
    size_t idx = (size_t)blockIdx.x * blockDim.x + threadIdx.x;
    if (idx >= (size_t)NPAD * DIM) return;
    float v = tanhf(acc[idx]);
    x32[idx] = v;
    xh[idx]  = (f16)v;
}

// out[b, 1200:1712] = emb32[frame_list[b, gold_id[b]]]
__global__ void gather_out_kernel(const float* __restrict__ emb32,
                                  const int* __restrict__ frame_list,
                                  const int* __restrict__ gold_id,
                                  float* __restrict__ out)
{
    const int b = blockIdx.x;
    const int label = frame_list[b * 16 + gold_id[b]];
    const float4* src = (const float4*)(emb32 + (size_t)label * DIM);
    float4*       dst = (float4*)(out + (size_t)b * OUTW + FRAMES);
    dst[threadIdx.x] = src[threadIdx.x];   // 128 threads x float4 = 512 floats
}

// ---------------- host side ----------------
static inline size_t alignup(size_t x) { return (x + 255) & ~(size_t)255; }

extern "C" void kernel_launch(void* const* d_in, const int* in_sizes, int n_in,
                              void* d_out, int out_size, void* d_ws, size_t ws_size,
                              hipStream_t stream)
{
    (void)in_sizes; (void)n_in; (void)out_size; (void)ws_size;

    const float* target_span = (const float*)d_in[0];   // (32, 2048)
    const float* frame_emb   = (const float*)d_in[1];   // (1201, 512)
    const float* role_emb    = (const float*)d_in[2];   // (10001, 512)
    const float* rel_W0      = (const float*)d_in[3];   // (5, 512, 512)
    const float* rel_W1      = (const float*)d_in[4];   // (5, 512, 512)
    const float* span_W1     = (const float*)d_in[5];   // (2048, 2048)
    const float* span_b1     = (const float*)d_in[6];   // (2048,)
    const float* span_W2     = (const float*)d_in[7];   // (2048, 512)
    const float* span_b2     = (const float*)d_in[8];   // (512,)
    const float* fp_W1       = (const float*)d_in[9];   // (512, 512)
    const float* fp_b1       = (const float*)d_in[10];  // (512,)
    const float* fp_W2       = (const float*)d_in[11];  // (512, 512)
    const float* fp_b2       = (const float*)d_in[12];  // (512,)
    const float* adj_vals    = (const float*)d_in[13];  // (5, 100000)
    const int*   fe_ids      = (const int*)  d_in[14];  // (10001,)
    const int*   adj_rows    = (const int*)  d_in[15];  // (5, 100000)
    const int*   adj_cols    = (const int*)  d_in[16];  // (5, 100000)
    const int*   gold_id     = (const int*)  d_in[17];  // (32,)
    const int*   frame_list  = (const int*)  d_in[18];  // (32, 16)
    float* out = (float*)d_out;                          // (32, 1712)

    // ---- workspace layout (~98 MB) ----
    char* w = (char*)d_ws;
    auto take = [&](size_t bytes) { char* p = w; w += alignup(bytes); return p; };
    float* x32   = (float*)take((size_t)NPAD * DIM * 4);
    f16*   xh    = (f16*)  take((size_t)NPAD * DIM * 2);
    float* xl    = (float*)take((size_t)NPAD * DIM * 4);
    float* acc   = (float*)take((size_t)NPAD * DIM * 4);
    f16*   Wt0   = (f16*)  take((size_t)NREL * DIM * DIM * 2);
    f16*   Wt1   = (f16*)  take((size_t)NREL * DIM * DIM * 2);
    f16*   W1t   = (f16*)  take((size_t)2048 * 2048 * 2);
    f16*   W2t   = (f16*)  take((size_t)512 * 2048 * 2);
    f16*   fpW1t = (f16*)  take((size_t)DIM * DIM * 2);
    f16*   fpW2t = (f16*)  take((size_t)DIM * DIM * 2);
    f16*   span_h= (f16*)  take((size_t)BATCH * 2048 * 2);
    float* h1_32 = (float*)take((size_t)BATCH * 2048 * 4);
    f16*   h1_h  = (f16*)  take((size_t)BATCH * 2048 * 2);
    float* tn32  = (float*)take((size_t)BATCH * DIM * 4);
    f16*   tn_h  = (f16*)  take((size_t)BATCH * DIM * 2);
    float* h2_32 = (float*)take((size_t)BATCH * DIM * 4);
    f16*   h2_h  = (f16*)  take((size_t)BATCH * DIM * 2);
    float* Q32   = (float*)take((size_t)BATCH * DIM * 4);
    f16*   Q_h   = (f16*)  take((size_t)BATCH * DIM * 2);

    const int TPB = 256;
    auto blocks = [&](size_t n) { return (unsigned)((n + TPB - 1) / TPB); };
    // grid for wave tile height 16*MT, 8 waves per block, 64 cols per block
    auto gemm_grid = [&](int Mpad, int Npad, int MT) {
        int mtiles = Mpad / (16 * MT);
        return dim3((unsigned)((mtiles + 7) / 8), (unsigned)((Npad + 63) / 64));
    };

    // ---- weight conversion (f32 -> f16, transposed to N x K) ----
    for (int r = 0; r < NREL; ++r) {
        convT_kernel<<<blocks((size_t)DIM * DIM), TPB, 0, stream>>>(
            rel_W0 + (size_t)r * DIM * DIM, Wt0 + (size_t)r * DIM * DIM, DIM, DIM);
        convT_kernel<<<blocks((size_t)DIM * DIM), TPB, 0, stream>>>(
            rel_W1 + (size_t)r * DIM * DIM, Wt1 + (size_t)r * DIM * DIM, DIM, DIM);
    }
    convT_kernel<<<blocks((size_t)2048 * 2048), TPB, 0, stream>>>(span_W1, W1t, 2048, 2048);
    convT_kernel<<<blocks((size_t)2048 * 512),  TPB, 0, stream>>>(span_W2, W2t, 2048, 512);
    convT_kernel<<<blocks((size_t)DIM * DIM),   TPB, 0, stream>>>(fp_W1, fpW1t, DIM, DIM);
    convT_kernel<<<blocks((size_t)DIM * DIM),   TPB, 0, stream>>>(fp_W2, fpW2t, DIM, DIM);
    conv_kernel<<<blocks((size_t)BATCH * 2048), TPB, 0, stream>>>(
        target_span, span_h, (size_t)BATCH * 2048);

    // ---- node features ----
    build_x_kernel<<<blocks((size_t)NPAD * DIM), TPB, 0, stream>>>(
        frame_emb, role_emb, fe_ids, x32, xh);

    // ---- two RelGCN layers ----
    for (int layer = 0; layer < 2; ++layer) {
        const f16* Wt = (layer == 0) ? Wt0 : Wt1;
        zero_kernel<<<blocks((size_t)NPAD * DIM), TPB, 0, stream>>>(acc, (size_t)NPAD * DIM);
        for (int r = 0; r < NREL; ++r) {
            wmma_gemm_kernel<0, 2><<<gemm_grid(NPAD, DIM, 2), TPB, 0, stream>>>(
                xh, Wt + (size_t)r * DIM * DIM, xl, (f16*)nullptr,
                (const float*)nullptr, NPAD, DIM, DIM, DIM);
            edge_scatter_kernel<<<NEDGE, 128, 0, stream>>>(
                xl, adj_rows + (size_t)r * NEDGE, adj_cols + (size_t)r * NEDGE,
                adj_vals + (size_t)r * NEDGE, acc);
        }
        tanh_kernel<<<blocks((size_t)NPAD * DIM), TPB, 0, stream>>>(acc, x32, xh);
    }
    // now x32 / xh hold all_embeddings (f32 / f16)

    // ---- MLP head (M = 32 -> MT = 1) ----
    wmma_gemm_kernel<1, 1><<<gemm_grid(32, 2048, 1), TPB, 0, stream>>>(
        span_h, W1t, h1_32, h1_h, span_b1, BATCH, 2048, 2048, 2048);
    wmma_gemm_kernel<0, 1><<<gemm_grid(32, DIM, 1), TPB, 0, stream>>>(
        h1_h, W2t, tn32, tn_h, span_b2, BATCH, DIM, 2048, DIM);
    wmma_gemm_kernel<1, 1><<<gemm_grid(32, DIM, 1), TPB, 0, stream>>>(
        tn_h, fpW1t, h2_32, h2_h, fp_b1, BATCH, DIM, DIM, DIM);
    wmma_gemm_kernel<2, 1><<<gemm_grid(32, DIM, 1), TPB, 0, stream>>>(
        h2_h, fpW2t, Q32, Q_h, fp_b2, BATCH, DIM, DIM, DIM);

    // ---- pred_frame_weight = Q @ emb[:1200]^T written straight into d_out ----
    // N padded 1200 -> 1216; the extra 16 columns land in d_out[:, 1200:1216],
    // which gather_out_kernel overwrites afterwards.
    wmma_gemm_kernel<0, 1><<<gemm_grid(32, 1216, 1), TPB, 0, stream>>>(
        Q_h, xh, out, (f16*)nullptr, (const float*)nullptr,
        BATCH, 1216, DIM, OUTW);

    // ---- frame_node gather into d_out[:, 1200:] ----
    gather_out_kernel<<<BATCH, 128, 0, stream>>>(x32, frame_list, gold_id, out);
}
